// DecoderLayer_83141976916908
// MI455X (gfx1250) — matrix operand order repaired and checked by reading, not compiled, not run on hardware
//
#include <hip/hip_runtime.h>
#include <hip/hip_bf16.h>

// ---------------------------------------------------------------------------
// Decoder layer on MI455X (gfx1250). All GEMMs + both attentions on
// v_wmma_f32_16x16x32_bf16 (wave32), f32 accumulate. Flash-style attention
// (online softmax) avoids materializing B*H*S*S scores (256MB > 192MB L2).
// GEMM: 128x128 block tile, double-buffered LDS staged via gfx1250 async
// global->LDS DMA (ASYNCcnt), so DMA of step i+1 overlaps WMMA of step i.
// All WMMA fragments are contiguous 16B chunks (2x b128 per operand).
// ---------------------------------------------------------------------------

typedef __attribute__((ext_vector_type(16))) __bf16 v16bf;
typedef __attribute__((ext_vector_type(8)))  __bf16 v8bf;
typedef __attribute__((ext_vector_type(8)))  float  v8f;

#define BQ 4
#define SQ 1024
#define DQ 1024
#define HQ 16
#define HIDQ 2048
#define MQ (BQ * SQ)
#define LN_EPS 1e-5f

#define SHUF16(lo, hi) \
  __builtin_shufflevector((lo), (hi), 0, 1, 2, 3, 4, 5, 6, 7, 8, 9, 10, 11, 12, 13, 14, 15)

static __device__ __forceinline__ void lds_fence() {
  asm volatile("s_wait_dscnt 0" ::: "memory");
}

// gfx1250 async global->LDS DMA (CDNA5 ISA): 16B per lane,
// tracked on ASYNCcnt. First operand = per-lane LDS byte offset VGPR.
static __device__ __forceinline__ void async_ld16(unsigned lds_off,
                                                  const void* gaddr) {
  asm volatile("global_load_async_to_lds_b128 %0, %1, off"
               :: "v"(lds_off), "v"(gaddr) : "memory");
}
static __device__ __forceinline__ void wait_async() {
  asm volatile("s_wait_asynccnt 0x0" ::: "memory");
}

// ---------------------------------------------------------------------------
// f32 -> bf16 convert (grid-stride)
// ---------------------------------------------------------------------------
__global__ void cvt_bf16_kernel(const float* __restrict__ in,
                                __bf16* __restrict__ out, int n) {
  for (int i = blockIdx.x * blockDim.x + threadIdx.x; i < n;
       i += gridDim.x * blockDim.x)
    out[i] = (__bf16)in[i];
}

// ---------------------------------------------------------------------------
// f32 [K][N] -> bf16 transposed [N][K]  (32x32 tiles, coalesced both sides)
// ---------------------------------------------------------------------------
__global__ void cvtT_bf16_kernel(const float* __restrict__ in,
                                 __bf16* __restrict__ out, int K, int N) {
  __shared__ float tile[32][33];
  const int k0 = blockIdx.y * 32, n0 = blockIdx.x * 32;
  const int tx = threadIdx.x & 31, ty = threadIdx.x >> 5;  // 32 x 8
  for (int i = 0; i < 32; i += 8)
    tile[ty + i][tx] = in[(size_t)(k0 + ty + i) * N + n0 + tx];
  __syncthreads();
  for (int i = 0; i < 32; i += 8)
    out[(size_t)(n0 + ty + i) * K + k0 + tx] = (__bf16)tile[tx][ty + i];
}

// ---------------------------------------------------------------------------
// Tiled bf16 GEMM:  C[M,N] = A[M,K] * W[K,N] + bias (W passed TRANSPOSED as
// WT[N][K]).  Block = 256 thr (8 waves), tile 128x128, BK=32, double-buffered
// async-DMA staging.  Wave (wm,wn) computes 32x64 via 8 wmma/K-step
// (2 A-frags x 4 B-frags, B reused x2, A reused x4).
// vmode: 1 -> also scatter V-part of qkv (N=3072) into vT[b][h][dk][s]
//        2 -> also scatter V-part of kv  (N=2048) into vT[b][h][dk][s]
// ---------------------------------------------------------------------------
__global__ void gemm_bf16_kernel(const __bf16* __restrict__ A,
                                 const __bf16* __restrict__ WT,
                                 const float* __restrict__ bias,
                                 float* __restrict__ outF,
                                 __bf16* __restrict__ outB,
                                 __bf16* __restrict__ vT,
                                 int M, int N, int K, int relu, int vmode) {
  // 128 x 32 (+8 pad) bf16, 80B rows; x2 ping-pong buffers
  __shared__ alignas(16) __bf16 As[2][128][40];
  __shared__ alignas(16) __bf16 BsT[2][128][40];
  const unsigned BUFB = 128 * 40 * 2;  // bytes per buffer

  const int tid  = threadIdx.x;
  const int lane = tid & 31;
  const int wave = tid >> 5;
  const int wm = wave & 3;          // 32-row group
  const int wn = wave >> 2;         // 64-col group
  const int m0 = blockIdx.y * 128;
  const int n0 = blockIdx.x * 128;

  const int hi = (lane >> 4) & 1;
  const int ml = lane & 15;
  const int kadd = hi ? 16 : 0;     // A-frag K base within 32-chunk
  const int koff = hi ? 16 : 0;     // B-frag K base within 32-chunk

  // staging coordinates: 2 chunks of 16B for A and 2 for B per thread
  const int r1 = tid >> 2,         c1 = (tid & 3) * 8;
  const int r2 = (tid + 256) >> 2, c2 = (tid & 3) * 8;
  const unsigned as0 = (unsigned)(size_t)&As[0][0][0];
  const unsigned bs0 = (unsigned)(size_t)&BsT[0][0][0];

  auto stage = [&](int buf, int k0) {
    const unsigned ab = as0 + (unsigned)buf * BUFB;
    const unsigned bb = bs0 + (unsigned)buf * BUFB;
    async_ld16(ab + (unsigned)(r1 * 80 + c1 * 2), A + (size_t)(m0 + r1) * K + k0 + c1);
    async_ld16(ab + (unsigned)(r2 * 80 + c2 * 2), A + (size_t)(m0 + r2) * K + k0 + c2);
    async_ld16(bb + (unsigned)(r1 * 80 + c1 * 2), WT + (size_t)(n0 + r1) * K + k0 + c1);
    async_ld16(bb + (unsigned)(r2 * 80 + c2 * 2), WT + (size_t)(n0 + r2) * K + k0 + c2);
  };

  v8f acc[2][4];
  for (int u = 0; u < 2; ++u)
    for (int t = 0; t < 4; ++t)
      for (int r = 0; r < 8; ++r) acc[u][t][r] = 0.0f;

  const int nsteps = K / 32;
  stage(0, 0);
  wait_async();
  __syncthreads();

  for (int i = 0; i < nsteps; ++i) {
    const int buf = i & 1;
    if (i + 1 < nsteps) stage(1 - buf, (i + 1) * 32);  // DMA overlaps WMMA

    const __bf16* a0 = &As[buf][wm * 32 + ml][kadd];
    const __bf16* a1 = &As[buf][wm * 32 + 16 + ml][kadd];
    v16bf af0 = SHUF16(*(const v8bf*)a0, *(const v8bf*)(a0 + 8));
    v16bf af1 = SHUF16(*(const v8bf*)a1, *(const v8bf*)(a1 + 8));
    for (int t = 0; t < 4; ++t) {
      const __bf16* bp = &BsT[buf][wn * 64 + t * 16 + ml][koff];
      v16bf bfr = SHUF16(*(const v8bf*)bp, *(const v8bf*)(bp + 8));
      acc[0][t] = __builtin_amdgcn_wmma_f32_16x16x32_bf16(
          false, af0, false, bfr, (short)0, acc[0][t], false, false);
      acc[1][t] = __builtin_amdgcn_wmma_f32_16x16x32_bf16(
          false, af1, false, bfr, (short)0, acc[1][t], false, false);
    }
    wait_async();
    __syncthreads();
  }

  // epilogue: bias (+ReLU); write f32 / bf16 / transposed-V copies
  for (int u = 0; u < 2; ++u)
    for (int t = 0; t < 4; ++t) {
      const int col = n0 + wn * 64 + t * 16 + ml;
      const float bv = bias ? bias[col] : 0.0f;
      for (int r = 0; r < 8; ++r) {
        const int row = m0 + wm * 32 + u * 16 + r + 8 * hi;
        float v = acc[u][t][r] + bv;
        if (relu) v = fmaxf(v, 0.0f);
        const size_t idx = (size_t)row * N + col;
        if (outF) outF[idx] = v;
        if (outB) outB[idx] = (__bf16)v;
        if (vmode == 1) {                       // qkv: 192 feats/head, V last 64
          int h = col / 192, rem = col - h * 192;
          if (rem >= 128) {
            int dk = rem - 128, b = row >> 10, s = row & 1023;
            vT[(((size_t)b * HQ + h) * 64 + dk) * SQ + s] = (__bf16)v;
          }
        } else if (vmode == 2) {                // kv: 128 feats/head, V last 64
          int h = col >> 7, rem = col & 127;
          if (rem >= 64) {
            int dk = rem - 64, b = row >> 10, s = row & 1023;
            vT[(((size_t)b * HQ + h) * 64 + dk) * SQ + s] = (__bf16)v;
          }
        }
      }
    }
}

// ---------------------------------------------------------------------------
// Flash attention, one wave per (b, h, 16-row query tile).
// q/k addressed as ptr[row*rs + h*hs + ho + d]; V pre-transposed vT[b][h][dk][s].
// out merged [M, D] bf16 at feature h*64 + d.
// ---------------------------------------------------------------------------
__global__ void attn_kernel(const __bf16* __restrict__ Qp, int q_rs, int q_hs, int q_ho,
                            const __bf16* __restrict__ Kp, int k_rs, int k_hs, int k_ho,
                            const __bf16* __restrict__ vT,
                            const float* __restrict__ mask,
                            __bf16* __restrict__ out, int causal) {
  __shared__ float sS[4][16][33];
  __shared__ alignas(16) __bf16 sP[4][16][40];
  __shared__ float sR[4][16];

  const int lane = threadIdx.x & 31;
  const int wave = threadIdx.x >> 5;
  const int task = blockIdx.x * 4 + wave;   // B*H*(S/16) = 4096 tasks
  const int qt = task & 63;
  const int h  = (task >> 6) & 15;
  const int b  = task >> 10;
  const int qrow0 = b * SQ + qt * 16;

  const int hi = (lane >> 4) & 1;
  const int ml = lane & 15;
  const int kadd = hi ? 16 : 0;
  const int koff = hi ? 16 : 0;

  // Q fragments (A-layout 16x32 each; K = 0..31 and 32..63), vectorized
  v16bf q0, q1;
  {
    const __bf16* qb = Qp + (size_t)(qrow0 + ml) * q_rs + h * q_hs + q_ho;
    q0 = SHUF16(*(const v8bf*)(qb + kadd), *(const v8bf*)(qb + kadd + 8));
    q1 = SHUF16(*(const v8bf*)(qb + kadd + 32), *(const v8bf*)(qb + kadd + 40));
  }

  v8f O[4];
  for (int t = 0; t < 4; ++t)
    for (int r = 0; r < 8; ++r) O[t][r] = 0.0f;

  float rm = -INFINITY, rl = 0.0f;   // lanes 0..15 own row `lane`
  const int nchunks = causal ? ((qt * 16 + 15) / 32 + 1) : (SQ / 32);
  const float scale = 0.125f;        // 1/sqrt(64)

  for (int c = 0; c < nchunks; ++c) {
    const int kb = c * 32;

    // --- scores: two 16x16 subtiles of Q(16x64) x K^T(64x16)
    for (int sub = 0; sub < 2; ++sub) {
      v8f s;
      for (int r = 0; r < 8; ++r) s[r] = 0.0f;
      const int key = b * SQ + kb + sub * 16 + ml;
      const __bf16* kbp = Kp + (size_t)key * k_rs + h * k_hs + k_ho;
      v16bf kf0 = SHUF16(*(const v8bf*)(kbp + koff), *(const v8bf*)(kbp + koff + 8));
      v16bf kf1 = SHUF16(*(const v8bf*)(kbp + koff + 32), *(const v8bf*)(kbp + koff + 40));
      s = __builtin_amdgcn_wmma_f32_16x16x32_bf16(false, q0, false, kf0,
                                                  (short)0, s, false, false);
      s = __builtin_amdgcn_wmma_f32_16x16x32_bf16(false, q1, false, kf1,
                                                  (short)0, s, false, false);
      for (int r = 0; r < 8; ++r)
        sS[wave][r + 8 * hi][sub * 16 + ml] = s[r];
    }
    lds_fence();

    // --- online softmax: lane r (<16) owns tile row r
    if (lane < 16) {
      float sc[32], mx = rm;
      for (int j = 0; j < 32; ++j) {
        float v = sS[wave][lane][j] * scale;
        if (mask) v += mask[(size_t)(qt * 16 + lane) * SQ + kb + j];
        sc[j] = v;
        mx = fmaxf(mx, v);
      }
      float sf = __expf(rm - mx);
      float ps = 0.0f;
      for (int j = 0; j < 32; ++j) {
        float p = __expf(sc[j] - mx);
        ps += p;
        sP[wave][lane][j] = (__bf16)p;
      }
      rl = rl * sf + ps;
      rm = mx;
      sR[wave][lane] = sf;
    }
    lds_fence();

    // --- rescale O, then O += P(16x32) x V(32x64)
    float sfv[8];
    for (int r = 0; r < 8; ++r) sfv[r] = sR[wave][r + 8 * hi];
    for (int t = 0; t < 4; ++t)
      for (int r = 0; r < 8; ++r) O[t][r] *= sfv[r];

    const __bf16* pp = &sP[wave][ml][0];
    v16bf pf = SHUF16(*(const v8bf*)(pp + kadd), *(const v8bf*)(pp + kadd + 8));

    for (int t = 0; t < 4; ++t) {
      const __bf16* vb = vT + (((size_t)b * HQ + h) * 64 + t * 16 + ml) * SQ
                            + kb + 16 * hi;
      v16bf vf = SHUF16(*(const v8bf*)vb, *(const v8bf*)(vb + 8));
      O[t] = __builtin_amdgcn_wmma_f32_16x16x32_bf16(false, pf, false, vf,
                                                     (short)0, O[t], false, false);
    }
    lds_fence();
  }

  // --- normalize by row sum, write merged bf16 output
  if (lane < 16) sR[wave][lane] = (rl > 0.0f) ? (1.0f / rl) : 0.0f;
  lds_fence();
  float inv[8];
  for (int r = 0; r < 8; ++r) inv[r] = sR[wave][r + 8 * hi];
  for (int t = 0; t < 4; ++t)
    for (int r = 0; r < 8; ++r) {
      int row = qrow0 + r + 8 * hi;
      out[(size_t)row * DQ + h * 64 + t * 16 + ml] = (__bf16)(O[t][r] * inv[r]);
    }
}

// ---------------------------------------------------------------------------
// LayerNorm over last dim (D=1024), one block (256 thr) per row.
//   base = a (+ pre)  ->  LN(base)*g + b (+ post) ; writes f32 and opt bf16.
// ---------------------------------------------------------------------------
__global__ void ln_kernel(const float* __restrict__ a,
                          const float* __restrict__ pre,
                          const float* __restrict__ post,
                          const float* __restrict__ g,
                          const float* __restrict__ bta,
                          float* __restrict__ outF,
                          __bf16* __restrict__ outB) {
  __shared__ float red[256];
  const int row = blockIdx.x;
  const int tid = threadIdx.x;
  const size_t base = (size_t)row * DQ;

  float v[4];
  float s = 0.0f, s2 = 0.0f;
  for (int j = 0; j < 4; ++j) {
    int idx = tid + j * 256;
    float t = a[base + idx] + (pre ? pre[base + idx] : 0.0f);
    v[j] = t;
    s += t;
    s2 += t * t;
  }
  red[tid] = s;
  __syncthreads();
  for (int off = 128; off > 0; off >>= 1) {
    if (tid < off) red[tid] += red[tid + off];
    __syncthreads();
  }
  const float mu = red[0] * (1.0f / DQ);
  __syncthreads();
  red[tid] = s2;
  __syncthreads();
  for (int off = 128; off > 0; off >>= 1) {
    if (tid < off) red[tid] += red[tid + off];
    __syncthreads();
  }
  const float var = red[0] * (1.0f / DQ) - mu * mu;
  const float rstd = rsqrtf(var + LN_EPS);

  for (int j = 0; j < 4; ++j) {
    int idx = tid + j * 256;
    float o = (v[j] - mu) * rstd * g[idx] + bta[idx];
    if (post) o += post[base + idx];
    if (outF) outF[base + idx] = o;
    if (outB) outB[base + idx] = (__bf16)o;
  }
}

// ---------------------------------------------------------------------------
// Host launcher
// ---------------------------------------------------------------------------
extern "C" void kernel_launch(void* const* d_in, const int* in_sizes, int n_in,
                              void* d_out, int out_size, void* d_ws, size_t ws_size,
                              hipStream_t stream) {
  (void)in_sizes; (void)n_in; (void)out_size; (void)ws_size;

  const float* x      = (const float*)d_in[0];
  const float* y      = (const float*)d_in[1];
  const float* mask   = (const float*)d_in[2];
  const float* qkv_w  = (const float*)d_in[3];
  const float* qkv_b  = (const float*)d_in[4];
  const float* ao_w   = (const float*)d_in[5];
  const float* ao_b   = (const float*)d_in[6];
  const float* kv_w   = (const float*)d_in[7];
  const float* kv_b   = (const float*)d_in[8];
  const float* q_w    = (const float*)d_in[9];
  const float* q_b    = (const float*)d_in[10];
  const float* co_w   = (const float*)d_in[11];
  const float* co_b   = (const float*)d_in[12];
  const float* fc1_w  = (const float*)d_in[13];
  const float* fc1_b  = (const float*)d_in[14];
  const float* fc2_w  = (const float*)d_in[15];
  const float* fc2_b  = (const float*)d_in[16];
  const float* ln1_g  = (const float*)d_in[17];
  const float* ln1_b  = (const float*)d_in[18];
  const float* ln2_g  = (const float*)d_in[19];
  const float* ln2_b  = (const float*)d_in[20];
  const float* ln3_g  = (const float*)d_in[21];
  const float* ln3_b  = (const float*)d_in[22];
  float* out = (float*)d_out;

  char* p = (char*)d_ws;
  auto alloc = [&](size_t bytes) -> void* {
    void* r = (void*)p;
    p += (bytes + 255) & ~(size_t)255;
    return r;
  };
  const size_t MD = (size_t)MQ * DQ;
  __bf16* ybf      = (__bf16*)alloc(MD * 2);
  __bf16* xbf      = (__bf16*)alloc(MD * 2);
  __bf16* wqkvT    = (__bf16*)alloc((size_t)DQ * 3 * DQ * 2);   // [3D][D]
  __bf16* waoT     = (__bf16*)alloc((size_t)DQ * DQ * 2);
  __bf16* wkvT     = (__bf16*)alloc((size_t)DQ * 2 * DQ * 2);
  __bf16* wqT      = (__bf16*)alloc((size_t)DQ * DQ * 2);
  __bf16* wcoT     = (__bf16*)alloc((size_t)DQ * DQ * 2);
  __bf16* wfc1T    = (__bf16*)alloc((size_t)DQ * HIDQ * 2);     // [HID][D]
  __bf16* wfc2T    = (__bf16*)alloc((size_t)HIDQ * DQ * 2);     // [D][HID]
  __bf16* qkv_bf   = (__bf16*)alloc(MD * 3 * 2);
  __bf16* vT_self  = (__bf16*)alloc(MD * 2);                    // [B,H,64,S]
  __bf16* attn_bf  = (__bf16*)alloc(MD * 2);
  float*  sa_f     = (float*)alloc(MD * 4);
  float*  yatt_f   = (float*)alloc(MD * 4);
  __bf16* kvx_bf   = (__bf16*)alloc(MD * 2 * 2);
  __bf16* vT_cross = (__bf16*)alloc(MD * 2);
  __bf16* cq_bf    = (__bf16*)alloc(MD * 2);
  __bf16* cross_bf = (__bf16*)alloc(MD * 2);
  float*  co_f     = (float*)alloc(MD * 4);
  float*  ycross_f = (float*)alloc(MD * 4);
  __bf16* ycross_bf= (__bf16*)alloc(MD * 2);
  __bf16* ff1_bf   = (__bf16*)alloc((size_t)MQ * HIDQ * 2);
  float*  ff2_f    = (float*)alloc(MD * 4);

  auto cvt = [&](const float* src, __bf16* dst, int n) {
    cvt_bf16_kernel<<<2048, 256, 0, stream>>>(src, dst, n);
  };
  auto cvtT = [&](const float* src, __bf16* dst, int K, int N) {
    cvtT_bf16_kernel<<<dim3(N / 32, K / 32), 256, 0, stream>>>(src, dst, K, N);
  };
  auto gemm = [&](const __bf16* A, const __bf16* WT, const float* bias,
                  float* oF, __bf16* oB, __bf16* vT, int M, int N, int K,
                  int relu, int vmode) {
    gemm_bf16_kernel<<<dim3(N / 128, M / 128), 256, 0, stream>>>(
        A, WT, bias, oF, oB, vT, M, N, K, relu, vmode);
  };

  // conversions: activations row-major, weights transposed [N][K]
  cvt(y, ybf, MQ * DQ);
  cvt(x, xbf, MQ * DQ);
  cvtT(qkv_w, wqkvT, DQ, 3 * DQ);
  cvtT(ao_w,  waoT,  DQ, DQ);
  cvtT(kv_w,  wkvT,  DQ, 2 * DQ);
  cvtT(q_w,   wqT,   DQ, DQ);
  cvtT(co_w,  wcoT,  DQ, DQ);
  cvtT(fc1_w, wfc1T, DQ, HIDQ);
  cvtT(fc2_w, wfc2T, HIDQ, DQ);

  // --- self attention on y (causal) ---
  gemm(ybf, wqkvT, qkv_b, nullptr, qkv_bf, vT_self, MQ, 3 * DQ, DQ, 0, 1);
  attn_kernel<<<1024, 128, 0, stream>>>(
      qkv_bf, 3 * DQ, 192, 0,        // q
      qkv_bf, 3 * DQ, 192, 64,       // k
      vT_self, mask, attn_bf, 1);
  gemm(attn_bf, waoT, ao_b, sa_f, nullptr, nullptr, MQ, DQ, DQ, 0, 0);
  ln_kernel<<<MQ, 256, 0, stream>>>(sa_f, y, nullptr, ln1_g, ln1_b,
                                    yatt_f, nullptr);

  // --- cross attention: kv from x, q from ORIGINAL y, no mask ---
  gemm(xbf, wkvT, kv_b, nullptr, kvx_bf, vT_cross, MQ, 2 * DQ, DQ, 0, 2);
  gemm(ybf, wqT, q_b, nullptr, cq_bf, nullptr, MQ, DQ, DQ, 0, 0);
  attn_kernel<<<1024, 128, 0, stream>>>(
      cq_bf, DQ, 64, 0,              // q
      kvx_bf, 2 * DQ, 128, 0,        // k
      vT_cross, nullptr, cross_bf, 0);
  gemm(cross_bf, wcoT, co_b, co_f, nullptr, nullptr, MQ, DQ, DQ, 0, 0);
  // quirk: y_cross = LN(cross) + y_att
  ln_kernel<<<MQ, 256, 0, stream>>>(co_f, nullptr, yatt_f, ln2_g, ln2_b,
                                    ycross_f, ycross_bf);

  // --- feedforward ---
  gemm(ycross_bf, wfc1T, fc1_b, nullptr, ff1_bf, nullptr, MQ, HIDQ, DQ, 1, 0);
  gemm(ff1_bf, wfc2T, fc2_b, ff2_f, nullptr, nullptr, MQ, DQ, HIDQ, 0, 0);
  ln_kernel<<<MQ, 256, 0, stream>>>(ff2_f, ycross_f, nullptr, ln3_g, ln3_b,
                                    out, nullptr);
}
